// CTGRU_Model_40647570489630
// MI455X (gfx1250) — matrix operand order repaired and checked by reading, not compiled, not run on hardware
//
#include <hip/hip_runtime.h>

#define BATCH   2048
#define TSTEPS  16
#define FDIM    512
#define UDIM    512
#define MDIM    8
#define OUTD    3
#define KTOT    1024
#define KC      32          // K chunks of 32 (KTOT/32)
#define KCH     16          // half: x-part chunks / h-part chunks
#define TB      16          // batch rows per workgroup
#define XPITCH  520         // row pitch (bf16 elems) = 512 + 8 pad (bank stagger)
#define NTHREADS 256        // 8 waves

typedef __attribute__((ext_vector_type(16))) __bf16        v16bf;
typedef __attribute__((ext_vector_type(8)))  float          v8f;
typedef __attribute__((ext_vector_type(16))) unsigned short u16x16;
typedef __attribute__((ext_vector_type(8)))  unsigned short u16x8;
typedef __attribute__((ext_vector_type(4)))  unsigned short u16x4;
typedef __attribute__((ext_vector_type(4)))  float          f32x4;

__device__ __forceinline__ unsigned short f2bf(float f) {
    unsigned u = __float_as_uint(f);
    u += 0x7FFFu + ((u >> 16) & 1u);      // round-to-nearest-even
    return (unsigned short)(u >> 16);
}
__device__ __forceinline__ float bf2f(unsigned short h) {
    return __uint_as_float(((unsigned)h) << 16);
}

__device__ __forceinline__ v8f wmma_bf16(u16x16 a, u16x16 b, v8f c) {
    union { u16x16 u; v16bf v; } ua, ub;
    ua.u = a; ub.u = b;
    return __builtin_amdgcn_wmma_f32_16x16x32_bf16(
        /*neg_a=*/false, ua.v, /*neg_b=*/false, ub.v,
        /*c_mod=*/(short)0, c, /*reuse_a=*/false, /*reuse_b=*/false);
}

// A fragment (16x32 bf16) from a single LDS region (no runtime select):
// lane row = lane&15; K = {half*8..+7} ∪ {16+half*8..+7}, half = lane>>4.
__device__ __forceinline__ u16x16 load_a_frag(const unsigned short* src,
                                              int kc, int row, int half) {
    const unsigned short* p = src + row * XPITCH + kc * 32 + half * 8;
    u16x8 lo = *(const u16x8*)(p);        // ds_load_b128
    u16x8 hi = *(const u16x8*)(p + 16);   // ds_load_b128
    return __builtin_shufflevector(lo, hi, 0,1,2,3,4,5,6,7,8,9,10,11,12,13,14,15);
}

// B fragment (32x16 bf16): each lane reads 32 contiguous bytes.
__device__ __forceinline__ u16x16 load_b_frag(const unsigned short* Wp,
                                              int nt, int kc, int lane) {
    return *(const u16x16*)(Wp + ((((size_t)nt * KC + kc) * 32 + lane) << 4));
}

__device__ __forceinline__ float red_max8(float v) {
    v = fmaxf(v, __shfl_xor(v, 1, 32));
    v = fmaxf(v, __shfl_xor(v, 2, 32));
    v = fmaxf(v, __shfl_xor(v, 4, 32));
    return v;
}
__device__ __forceinline__ float red_sum8(float v) {
    v += __shfl_xor(v, 1, 32);
    v += __shfl_xor(v, 2, 32);
    v += __shfl_xor(v, 4, 32);
    return v;
}

// Paired-tile GEMM: one shared A fragment feeds two independent WMMA chains.
// K loop split: x-region chunks (kc 0..15) then h-region chunks (kc 16..31).
__device__ __forceinline__ void gemm_pair(const unsigned short* sx,
                                          const unsigned short* sh,
                                          const unsigned short* Wp, int nt,
                                          int lane, int rowA, int half,
                                          v8f& acc0, v8f& acc1) {
    #pragma unroll 8
    for (int kc = 0; kc < KCH; ++kc) {
        u16x16 a  = load_a_frag(sx, kc, rowA, half);
        u16x16 b0 = load_b_frag(Wp, nt,     kc, lane);
        u16x16 b1 = load_b_frag(Wp, nt + 1, kc, lane);
        acc0 = wmma_bf16(a, b0, acc0);
        acc1 = wmma_bf16(a, b1, acc1);
    }
    #pragma unroll 8
    for (int kc = 0; kc < KCH; ++kc) {
        u16x16 a  = load_a_frag(sh, kc, rowA, half);
        u16x16 b0 = load_b_frag(Wp, nt,     KCH + kc, lane);
        u16x16 b1 = load_b_frag(Wp, nt + 1, KCH + kc, lane);
        acc0 = wmma_bf16(a, b0, acc0);
        acc1 = wmma_bf16(a, b1, acc1);
    }
}

// R epilogue for one tile: softmax over m (8-lane groups), ctx -> scr.
__device__ __forceinline__ void r_epilogue(v8f acc, float bias, int u, int half,
                                           int mlane, float ln_tau,
                                           const float* H, unsigned short* scr) {
    #pragma unroll
    for (int r = 0; r < 8; ++r) {
        float d  = (acc[r] + bias) - ln_tau;
        float ev = -d * d;
        float mx = red_max8(ev);
        float wv = __expf(ev - mx);
        float rn = wv / red_sum8(wv);
        int brow = r + half * 8;
        float ctx = rn * H[(brow * UDIM + u) * MDIM + mlane];
        ctx = red_sum8(ctx);
        if (mlane == 0) scr[brow * XPITCH + u] = f2bf(ctx);
    }
}

// S epilogue for one tile: softmax, state update, hidden -> scr.
__device__ __forceinline__ void s_epilogue(v8f acc, float bias, int u, int half,
                                           int mlane, int lane, float ln_tau,
                                           float decay, v8f qrow,
                                           float* H, unsigned short* scr) {
    int srcLane = (lane & 16) | (u & 15);
    #pragma unroll
    for (int r = 0; r < 8; ++r) {
        float qv = __shfl(qrow[r], srcLane, 32);
        float d  = (acc[r] + bias) - ln_tau;
        float ev = -d * d;
        float mx = red_max8(ev);
        float wv = __expf(ev - mx);
        float sn = wv / red_sum8(wv);
        int brow = r + half * 8;
        int hidx = (brow * UDIM + u) * MDIM + mlane;
        float hn = ((1.f - sn) * H[hidx] + sn * qv) * decay;
        H[hidx] = hn;
        float hs = red_sum8(hn);
        if (mlane == 0) scr[brow * XPITCH + u] = f2bf(hs);
    }
}

// Pack f32 KxN weight into WMMA-B fragment tiles (bf16):
// P[((nt*KC + kc)*32 + lane)*16 + e] = W[(kc*32 + (e>>3)*16 + (lane>>4)*8 + (e&7))*N + nt*16 + (lane&15)]
__global__ void pack_w(const float* __restrict__ W, unsigned short* __restrict__ P, int N) {
    int idx  = blockIdx.x * blockDim.x + threadIdx.x;
    int e    = idx & 15;
    int lane = (idx >> 4) & 31;
    int t    = idx >> 9;
    int kc   = t & (KC - 1);
    int nt   = t >> 5;
    int k    = kc * 32 + (e >> 3) * 16 + (lane >> 4) * 8 + (e & 7);
    int n    = nt * 16 + (lane & 15);
    P[idx] = f2bf(W[(size_t)k * N + n]);
}

__global__ __launch_bounds__(NTHREADS) void ctgru_main(
    const float* __restrict__ x,
    const unsigned short* __restrict__ WrP, const float* __restrict__ br,
    const unsigned short* __restrict__ WqP, const float* __restrict__ bq,
    const unsigned short* __restrict__ WsP, const float* __restrict__ bs,
    const float* __restrict__ Wo, const float* __restrict__ bo,
    float* __restrict__ gH, float* __restrict__ out)
{
    __shared__ __attribute__((aligned(16))) unsigned short s_x[TB * XPITCH];
    __shared__ __attribute__((aligned(16))) unsigned short s_bufA[TB * XPITCH];
    __shared__ __attribute__((aligned(16))) unsigned short s_bufB[TB * XPITCH];

    const int tid  = threadIdx.x;
    const int lane = tid & 31;
    const int wave = tid >> 5;
    const int b0   = blockIdx.x * TB;
    float* H = gH + (size_t)blockIdx.x * (TB * UDIM * MDIM);  // per-WG f32 state

    // Zero state + h buffer each launch (deterministic / graph-safe)
    for (int i = tid; i < TB * UDIM * MDIM; i += NTHREADS) H[i] = 0.f;
    for (int i = tid; i < TB * XPITCH;      i += NTHREADS) s_bufA[i] = 0;
    __threadfence_block();
    __syncthreads();

    const int   half  = lane >> 4;
    const int   rowA  = lane & 15;
    const int   colq  = lane & 15;
    const int   grp   = (lane >> 3) & 1;   // which u of the pair in an r/s tile
    const int   mlane = lane & 7;          // timescale index m for this lane
    const float ln_tau = (float)mlane * 1.1512925464970229f;       // m*0.5*ln(10)
    const float decay  = __expf(-0.04f / (__expf(ln_tau) + 1e-7f));

    unsigned short* hbuf = s_bufA;   // holds h_{t-1} (bf16)
    unsigned short* scr  = s_bufB;   // holds ctx, then hidden_t

    for (int t = 0; t < TSTEPS; ++t) {
        // ---- stage x_t (bf16), vectorized float4 -> 4x bf16 ----
        for (int i = tid; i < (TB * FDIM) / 4; i += NTHREADS) {
            int b  = i >> 7;                    // 128 quads per row
            int f4 = (i & 127) * 4;
            f32x4 v = *(const f32x4*)(x + ((size_t)(b0 + b) * TSTEPS + t) * FDIM + f4);
            u16x4 o;
            o[0] = f2bf(v[0]); o[1] = f2bf(v[1]);
            o[2] = f2bf(v[2]); o[3] = f2bf(v[3]);
            *(u16x4*)(s_x + b * XPITCH + f4) = o;
        }
        __syncthreads();

        // ---- R pass: ln_r = fi @ W_r ; softmax over m ; ctx -> scr ----
        for (int i = 0; i < 32; i += 2) {
            int nt = wave * 32 + i;                 // tile pair (nt, nt+1)
            v8f acc0 = {0.f,0.f,0.f,0.f,0.f,0.f,0.f,0.f};
            v8f acc1 = {0.f,0.f,0.f,0.f,0.f,0.f,0.f,0.f};
            gemm_pair(s_x, hbuf, WrP, nt, lane, rowA, half, acc0, acc1);
            r_epilogue(acc0, br[nt * 16 + colq],       nt * 2 + grp,       half, mlane, ln_tau, H, scr);
            r_epilogue(acc1, br[(nt + 1) * 16 + colq], (nt + 1) * 2 + grp, half, mlane, ln_tau, H, scr);
        }
        __syncthreads();

        // ---- Q pass: q = tanh([x|ctx] @ W_q + b_q), kept in registers ----
        v8f qf[4];
        #pragma unroll
        for (int jp = 0; jp < 2; ++jp) {
            int nt = wave * 4 + jp * 2;             // pair (nt, nt+1), 0..31
            v8f acc0 = {0.f,0.f,0.f,0.f,0.f,0.f,0.f,0.f};
            v8f acc1 = {0.f,0.f,0.f,0.f,0.f,0.f,0.f,0.f};
            gemm_pair(s_x, scr, WqP, nt, lane, rowA, half, acc0, acc1);
            float bias0 = bq[nt * 16 + colq];
            float bias1 = bq[(nt + 1) * 16 + colq];
            #pragma unroll
            for (int r = 0; r < 8; ++r) {
                float v0 = acc0[r] + bias0;
                float v1 = acc1[r] + bias1;
                float e0 = __expf(2.f * v0);
                float e1 = __expf(2.f * v1);
                qf[jp * 2][r]     = 1.f - 2.f / (e0 + 1.f);   // tanh
                qf[jp * 2 + 1][r] = 1.f - 2.f / (e1 + 1.f);
            }
        }
        __syncthreads();                             // ctx reads done; scr reusable

        // ---- S pass: ln_s = fi @ W_s ; softmax ; update state ; hidden -> scr ----
        #pragma unroll
        for (int j = 0; j < 4; ++j) {
            for (int ii = 0; ii < 8; ii += 2) {
                int nt = wave * 32 + j * 8 + ii;     // pair (nt, nt+1), same qf[j]
                v8f acc0 = {0.f,0.f,0.f,0.f,0.f,0.f,0.f,0.f};
                v8f acc1 = {0.f,0.f,0.f,0.f,0.f,0.f,0.f,0.f};
                gemm_pair(s_x, hbuf, WsP, nt, lane, rowA, half, acc0, acc1);
                s_epilogue(acc0, bs[nt * 16 + colq],       nt * 2 + grp,       half,
                           mlane, lane, ln_tau, decay, qf[j], H, scr);
                s_epilogue(acc1, bs[(nt + 1) * 16 + colq], (nt + 1) * 2 + grp, half,
                           mlane, lane, ln_tau, decay, qf[j], H, scr);
            }
        }
        __syncthreads();

        // ---- output projection: out[b,t,:] = hidden @ W_o + b_o ----
        if (tid < TB * OUTD) {
            int b = tid / OUTD, o = tid - b * OUTD;
            float accum = bo[o];
            for (int u2 = 0; u2 < UDIM; ++u2)
                accum += bf2f(scr[b * XPITCH + u2]) * Wo[u2 * OUTD + o];
            out[((size_t)(b0 + b) * TSTEPS + t) * OUTD + o] = accum;
        }

        // hidden becomes next step's h
        unsigned short* tmp = hbuf; hbuf = scr; scr = tmp;
        __syncthreads();
    }
}

extern "C" void kernel_launch(void* const* d_in, const int* in_sizes, int n_in,
                              void* d_out, int out_size, void* d_ws, size_t ws_size,
                              hipStream_t stream) {
    (void)in_sizes; (void)n_in; (void)out_size; (void)ws_size;
    const float* x   = (const float*)d_in[0];
    const float* W_r = (const float*)d_in[1];
    const float* b_r = (const float*)d_in[2];
    const float* W_q = (const float*)d_in[3];
    const float* b_q = (const float*)d_in[4];
    const float* W_s = (const float*)d_in[5];
    const float* b_s = (const float*)d_in[6];
    const float* W_o = (const float*)d_in[7];
    const float* b_o = (const float*)d_in[8];

    unsigned short* WrP = (unsigned short*)d_ws;
    unsigned short* WsP = WrP + (size_t)KTOT * 4096;
    unsigned short* WqP = WsP + (size_t)KTOT * 4096;
    float*          Hws = (float*)(WqP + (size_t)KTOT * 512);   // 2048*512*8 f32

    pack_w<<<(KTOT * 4096) / NTHREADS, NTHREADS, 0, stream>>>(W_r, WrP, 4096);
    pack_w<<<(KTOT * 4096) / NTHREADS, NTHREADS, 0, stream>>>(W_s, WsP, 4096);
    pack_w<<<(KTOT * 512)  / NTHREADS, NTHREADS, 0, stream>>>(W_q, WqP, 512);

    ctgru_main<<<BATCH / TB, NTHREADS, 0, stream>>>(
        x, WrP, b_r, WqP, b_q, WsP, b_s, W_o, b_o, Hws, (float*)d_out);
}
